// SpatialTemporalGNN_40252433498165
// MI455X (gfx1250) — matrix operand order repaired and checked by reading, not compiled, not run on hardware
//
#include <hip/hip_runtime.h>
#include <hip/hip_bf16.h>
#include <math.h>

// ---------------- problem constants ----------------
#define Bsz   16
#define Tt    50
#define NODES 62
#define DEG   8
#define INF_  64
#define Cc    128
#define Hh    256
#define NCc   3
#define Kk    5
#define Gg    (Bsz * Tt)            // 800 graphs
#define Nn    (Gg * NODES)          // 49600 nodes
#define Ee    (Gg * NODES * DEG)    // 396800 edges
#define EPG   (NODES * DEG)         // 496 edges per graph
#define EPG2  (EPG + NODES)         // 558 incl self loops

typedef __attribute__((ext_vector_type(16))) _Float16 v16h;
typedef __attribute__((ext_vector_type(8)))  _Float16 v8h;
typedef __attribute__((ext_vector_type(4)))  _Float16 v4h;
typedef __attribute__((ext_vector_type(8)))  float    v8f;
typedef int v4i_vs __attribute__((vector_size(16)));   // matches builtin param type

// -------- CDNA5 async global->LDS path (guarded; fallback = vector copy) ----
#if defined(__has_builtin)
# if __has_builtin(__builtin_amdgcn_global_load_async_to_lds_b128)
#  define ASYNC_LDS 1
# endif
#endif
#ifndef ASYNC_LDS
# define ASYNC_LDS 0
#endif

__device__ __forceinline__ void wait_async_zero() {
#if defined(__has_builtin) && __has_builtin(__builtin_amdgcn_s_wait_asynccnt)
  __builtin_amdgcn_s_wait_asynccnt(0);
#else
  asm volatile("s_wait_asynccnt 0" ::: "memory");
#endif
}

// ---------------- WMMA fragment helpers (ISA 7.12.2 layouts, wave32) -------
// A: 16x32 (MxK) f16 from row-major f16: lane<16 row=lane holds K {0..7,16..23},
// lane>=16 row=lane-16 holds K {8..15,24..31}. Two aligned 16B loads + shuffle.
__device__ __forceinline__ v16h load_A_f16(const _Float16* A, int lda, int lane) {
  const int row = lane & 15;
  const int kb  = (lane >> 4) << 3;
  const _Float16* p = A + row * lda + kb;
  v8h lo = *(const v8h*)(p);
  v8h hi = *(const v8h*)(p + 16);
  return __builtin_shufflevector(lo, hi, 0,1,2,3,4,5,6,7,8,9,10,11,12,13,14,15);
}

// B fragments are pre-packed to lane-major layout: one 32B contiguous chunk/lane
__device__ __forceinline__ v16h load_B_pk(const _Float16* P, int frag, int lane) {
  return *(const v16h*)(P + ((size_t)frag * 32 + lane) * 16);
}

#define WMMA_F16(a, b, c) \
  __builtin_amdgcn_wmma_f32_16x16x32_f16(false, (a), false, (b), (short)0, (c), false, false)

// ---------------- kernel 0a: f32 -> f16 convert (x staging) ----------------
__global__ void cvt_f16_kernel(const float* __restrict__ src,
                               _Float16* __restrict__ dst, int n4) {
  const int i = (blockIdx.x * 256 + threadIdx.x);
  if (i >= n4) return;
  const float4 v = *(const float4*)(src + (size_t)i * 4);
  v4h h; h[0] = (_Float16)v.x; h[1] = (_Float16)v.y; h[2] = (_Float16)v.z; h[3] = (_Float16)v.w;
  *(v4h*)(dst + (size_t)i * 4) = h;
}

// ---------------- kernel 0b: pack weight into WMMA-B fragment layout -------
// frag = kk*NT + nt ; lane layout per ISA: col = lane&15, K half = lane>>4.
__global__ void pack_w_kernel(const float* __restrict__ W, _Float16* __restrict__ dst,
                              int NT, int NKK, int ld, int transposed) {
  const int idx = blockIdx.x * 256 + threadIdx.x;
  const int lane = idx & 31, frag = idx >> 5;
  if (frag >= NT * NKK) return;
  const int kk = frag / NT, nt = frag % NT;
  const int col = lane & 15, kb = (lane >> 4) << 3;
  v16h b;
#pragma unroll
  for (int i = 0; i < 16; ++i) {
    const int k = kk * 32 + kb + i + (i >= 8 ? 8 : 0), n = nt * 16 + col;
    b[i] = (_Float16)(transposed ? W[(size_t)n * ld + k] : W[(size_t)k * ld + n]);
  }
  *(v16h*)(dst + ((size_t)frag * 32 + lane) * 16) = b;
}

// ---------------- kernel 1: xl = x@W_l+b_l, xr = x@W_r+b_r (WMMA) ----------
// M=49600 (3100 mtiles), K=64 (2 ksteps), N=128 (8 ntiles), x2 matrices.
__global__ void xform_kernel(const _Float16* __restrict__ xh,
                             const _Float16* __restrict__ pW_l, const float* __restrict__ b_l,
                             const _Float16* __restrict__ pW_r, const float* __restrict__ b_r,
                             float* __restrict__ xl, float* __restrict__ xr) {
  const int wave = threadIdx.x >> 5, lane = threadIdx.x & 31;
  const int tile = blockIdx.x * 8 + wave;            // 49600 tiles total
  const int which = tile / 24800;                    // 0 -> W_l, 1 -> W_r
  const int rem = tile % 24800;
  const int mt = rem >> 3, nt = rem & 7;
  const _Float16* pW = which ? pW_r : pW_l;
  const float* bias  = which ? b_r  : b_l;
  float*       out   = which ? xr   : xl;
  v8f c = {};
#pragma unroll
  for (int kk = 0; kk < 2; ++kk) {
    v16h a = load_A_f16(xh + (size_t)mt * 16 * INF_ + kk * 32, INF_, lane);
    v16h b = load_B_pk(pW, kk * 8 + nt, lane);
    c = WMMA_F16(a, b, c);
  }
  const int col = lane & 15, rb = (lane >> 4) << 3;
  const float bv = bias[nt * 16 + col];
#pragma unroll
  for (int j = 0; j < 8; ++j)
    out[((size_t)mt * 16 + rb + j) * Cc + nt * 16 + col] = c[j] + bv;
}

// ---------------- kernel 2: per-graph GATv2 attention + mean pool (LDS) ----
__device__ __forceinline__ void atomicMaxFloatLds(float* addr, float val) {
  int* ia = (int*)addr;
  int old = __float_as_int(*addr);
  while (__int_as_float(old) < val) {
    int assumed = old;
    old = atomicCAS(ia, assumed, __float_as_int(val));
    if (old == assumed) break;
  }
}

#define LDPAD 132   // multiple of 4 (16B-aligned rows for b128 async); 132%64=4

__global__ void gat_kernel(const float* __restrict__ xl, const float* __restrict__ xr,
                           const int* __restrict__ src, const int* __restrict__ dst,
                           const float* __restrict__ edge_attr,
                           const float* __restrict__ W_e, const float* __restrict__ att,
                           const float* __restrict__ bias_out,
                           _Float16* __restrict__ pooled16) {
  extern __shared__ float lds[];
  float* s_xl  = lds;                       // 62*132
  float* s_xr  = s_xl + NODES * LDPAD;      // 62*132 (aliased as h_acc later)
  float* s_e   = s_xr + NODES * LDPAD;      // 560
  float* s_we  = s_e + 560;                 // 128
  float* s_att = s_we + Cc;                 // 128
  float* s_la  = s_att + Cc;                // 62 (edge-attr sum -> loop_attr)
  float* s_cnt = s_la + NODES;              // 62
  float* s_mx  = s_cnt + NODES;             // 62
  float* s_dn  = s_mx + NODES;              // 62

  const int g = blockIdx.x;
  const int tid = threadIdx.x;
  const int ebase = g * EPG;
  const int noff = g * NODES;

  // stage xl/xr tiles into LDS: async DMA path on CDNA5, vector copy otherwise
  for (int i = tid; i < (NODES * Cc) / 4; i += 256) {
    const int nd = (i * 4) >> 7, cidx = (i * 4) & 127;
    const float* gl = xl + (size_t)(noff + nd) * Cc + cidx;
    const float* gr = xr + (size_t)(noff + nd) * Cc + cidx;
    float* dl = s_xl + nd * LDPAD + cidx;
    float* dr = s_xr + nd * LDPAD + cidx;
#if ASYNC_LDS
    __builtin_amdgcn_global_load_async_to_lds_b128((v4i_vs*)gl, (v4i_vs*)dl, 0, 0);
    __builtin_amdgcn_global_load_async_to_lds_b128((v4i_vs*)gr, (v4i_vs*)dr, 0, 0);
#else
    *(float4*)dl = *(const float4*)gl;
    *(float4*)dr = *(const float4*)gr;
#endif
  }
  for (int i = tid; i < Cc; i += 256) { s_we[i] = W_e[i]; s_att[i] = att[i]; }
  for (int i = tid; i < NODES; i += 256) {
    s_la[i] = 0.f; s_cnt[i] = 0.f; s_mx[i] = -3.4e38f; s_dn[i] = 0.f;
  }
#if ASYNC_LDS
  wait_async_zero();
#endif
  __syncthreads();

  // self-loop edge_attr = mean of incoming edge attrs (fill_value='mean')
  for (int e = tid; e < EPG; e += 256) {
    const int d = dst[ebase + e] - noff;
    atomicAdd(&s_la[d], edge_attr[ebase + e]);
    atomicAdd(&s_cnt[d], 1.0f);
  }
  __syncthreads();
  for (int i = tid; i < NODES; i += 256) s_la[i] = s_la[i] / fmaxf(s_cnt[i], 1.0f);
  __syncthreads();

  // e = leaky_relu(xl[s]+xr[d]+ea*W_e, 0.2) . att  ; running max per dst
  for (int e = tid; e < EPG2; e += 256) {
    int s, d; float ea;
    if (e < EPG) { s = src[ebase + e] - noff; d = dst[ebase + e] - noff; ea = edge_attr[ebase + e]; }
    else         { s = d = e - EPG; ea = s_la[s]; }
    float acc = 0.f;
    for (int cidx = 0; cidx < Cc; ++cidx) {
      float m = s_xl[s * LDPAD + cidx] + s_xr[d * LDPAD + cidx] + ea * s_we[cidx];
      m = m > 0.f ? m : 0.2f * m;
      acc += m * s_att[cidx];
    }
    s_e[e] = acc;
    atomicMaxFloatLds(&s_mx[d], acc);
  }
  __syncthreads();

  // scatter-softmax numerator + denominator
  for (int e = tid; e < EPG2; e += 256) {
    const int d = (e < EPG) ? dst[ebase + e] - noff : e - EPG;
    const float ez = expf(s_e[e] - s_mx[d]);
    s_e[e] = ez;
    atomicAdd(&s_dn[d], ez);
  }
  __syncthreads();

  // aggregate alpha * xl[src] into h_acc (reuse xr storage; xr no longer read)
  float* h_acc = s_xr;
  for (int i = tid; i < NODES * LDPAD; i += 256) h_acc[i] = 0.f;
  __syncthreads();
  for (int e = tid; e < EPG2; e += 256) {
    int s, d;
    if (e < EPG) { s = src[ebase + e] - noff; d = dst[ebase + e] - noff; }
    else         { s = d = e - EPG; }
    const float alpha = s_e[e] / s_dn[d];
    for (int cidx = 0; cidx < Cc; ++cidx)
      atomicAdd(&h_acc[d * LDPAD + cidx], alpha * s_xl[s * LDPAD + cidx]);
  }
  __syncthreads();

  // fused mean pool over the graph's nodes (+bias_out) -> f16 for the gx GEMM
  for (int cidx = tid; cidx < Cc; cidx += 256) {
    float acc = 0.f;
    for (int nd = 0; nd < NODES; ++nd) acc += h_acc[nd * LDPAD + cidx];
    pooled16[(size_t)g * Cc + cidx] = (_Float16)(acc * (1.0f / NODES) + bias_out[cidx]);
  }
}

// ---------------- kernel 3: gx_all = pooled @ W_ih^T (WMMA, batched) -------
// M=800 (50 mtiles), K=128 (4 ksteps), N=768 (48 ntiles)
__global__ void gx_kernel(const _Float16* __restrict__ pooled16,
                          const _Float16* __restrict__ pW_ih,
                          float* __restrict__ gx_all) {
  const int wave = threadIdx.x >> 5, lane = threadIdx.x & 31;
  const int tile = blockIdx.x * 8 + wave;   // 2400 tiles
  const int mt = tile / 48, nt = tile % 48;
  v8f c = {};
#pragma unroll
  for (int kk = 0; kk < 4; ++kk) {
    v16h a = load_A_f16(pooled16 + (size_t)mt * 16 * Cc + kk * 32, Cc, lane);
    v16h b = load_B_pk(pW_ih, kk * 48 + nt, lane);
    c = WMMA_F16(a, b, c);
  }
  const int col = lane & 15, rb = (lane >> 4) << 3;
#pragma unroll
  for (int j = 0; j < 8; ++j)
    gx_all[((size_t)mt * 16 + rb + j) * (3 * Hh) + nt * 16 + col] = c[j];
}

// ---------------- kernel 4: GRU recurrence (1 persistent block, WMMA) ------
// per step: gh = h @ W_hh^T (M=16, K=256 -> 8 ksteps, N=768 -> 48 ntiles, 6/wave)
__global__ void gru_kernel(const float* __restrict__ gx_all,
                           const _Float16* __restrict__ pW_hh,
                           const float* __restrict__ b_ih,
                           const float* __restrict__ b_hh,
                           _Float16* __restrict__ gru_out16) {
  extern __shared__ float lds[];
  float* gh  = lds;                              // 16*768 f32
  float* h32a = gh + Bsz * 3 * Hh;               // 16*256 f32
  float* h32b = h32a + Bsz * Hh;                 // 16*256 f32
  _Float16* h16a = (_Float16*)(h32b + Bsz * Hh); // 16*256 f16 (WMMA A mirror)
  _Float16* h16b = h16a + Bsz * Hh;              // 16*256 f16
  const int tid = threadIdx.x;
  const int wave = tid >> 5, lane = tid & 31;
  for (int i = tid; i < Bsz * Hh; i += 256) { h32a[i] = 0.f; h16a[i] = (_Float16)0.f; }
  __syncthreads();

  float*    hc32 = h32a; float*    hx32 = h32b;
  _Float16* hc16 = h16a; _Float16* hx16 = h16b;
  for (int t = 0; t < Tt; ++t) {
    for (int nt = wave * 6; nt < wave * 6 + 6; ++nt) {
      v8f c = {};
#pragma unroll
      for (int kk = 0; kk < 8; ++kk) {
        v16h a = load_A_f16(hc16 + kk * 32, Hh, lane);
        v16h b = load_B_pk(pW_hh, kk * 48 + nt, lane);
        c = WMMA_F16(a, b, c);
      }
      const int col = lane & 15, rb = (lane >> 4) << 3;
#pragma unroll
      for (int j = 0; j < 8; ++j)
        gh[(rb + j) * (3 * Hh) + nt * 16 + col] = c[j];
    }
    __syncthreads();

    // gate math: r,z,n (PyTorch order), h' = (1-z)*n + z*h
    for (int i = tid; i < Bsz * Hh; i += 256) {
      const int b = i >> 8, j = i & 255;
      const float* gxr = gx_all + (size_t)(b * Tt + t) * (3 * Hh);
      if (t + 1 < Tt) __builtin_prefetch(gxr + 3 * Hh + j, 0, 1);  // next step's gates
      const float xr_ = gxr[j]           + b_ih[j];
      const float xz  = gxr[Hh + j]      + b_ih[Hh + j];
      const float xn  = gxr[2 * Hh + j]  + b_ih[2 * Hh + j];
      const float hr  = gh[b * 3 * Hh + j]          + b_hh[j];
      const float hz  = gh[b * 3 * Hh + Hh + j]     + b_hh[Hh + j];
      const float hng = gh[b * 3 * Hh + 2 * Hh + j] + b_hh[2 * Hh + j];
      const float hprev = hc32[b * Hh + j];
      const float r = 1.f / (1.f + expf(-(xr_ + hr)));
      const float z = 1.f / (1.f + expf(-(xz + hz)));
      const float nn = tanhf(xn + r * hng);
      const float h2 = (1.f - z) * nn + z * hprev;
      hx32[b * Hh + j] = h2;
      hx16[b * Hh + j] = (_Float16)h2;
      gru_out16[(size_t)(b * Tt + t) * Hh + j] = (_Float16)h2;
    }
    __syncthreads();
    float* t32 = hc32; hc32 = hx32; hx32 = t32;
    _Float16* t16 = hc16; hc16 = hx16; hx16 = t16;
  }
}

// ---------------- kernel 5: hidden = relu(gru_out @ W1 + b1) (WMMA) --------
// M=800 (50 mtiles), K=256 (8 ksteps), N=128 (8 ntiles)
__global__ void hidden_kernel(const _Float16* __restrict__ gru_out16,
                              const _Float16* __restrict__ pW1,
                              const float* __restrict__ b1,
                              float* __restrict__ hidden) {
  const int wave = threadIdx.x >> 5, lane = threadIdx.x & 31;
  const int tile = blockIdx.x * 8 + wave;   // 400 tiles
  const int mt = tile >> 3, nt = tile & 7;
  v8f c = {};
#pragma unroll
  for (int kk = 0; kk < 8; ++kk) {
    v16h a = load_A_f16(gru_out16 + (size_t)mt * 16 * Hh + kk * 32, Hh, lane);
    v16h b = load_B_pk(pW1, kk * 8 + nt, lane);
    c = WMMA_F16(a, b, c);
  }
  const int col = lane & 15, rb = (lane >> 4) << 3;
  const float bv = b1[nt * 16 + col];
#pragma unroll
  for (int j = 0; j < 8; ++j)
    hidden[((size_t)mt * 16 + rb + j) * (Hh / 2) + nt * 16 + col] = fmaxf(c[j] + bv, 0.f);
}

// ---------------- kernel 6: scores = hidden @ W2 + b2 (tiny N=3) -----------
__global__ void scores_kernel(const float* __restrict__ hidden,
                              const float* __restrict__ W2,
                              const float* __restrict__ b2,
                              float* __restrict__ scores) {
  const int i = blockIdx.x * 256 + threadIdx.x;
  if (i >= Bsz * Tt * NCc) return;
  const int m = i / NCc, cidx = i % NCc;
  float acc = b2[cidx];
  for (int j = 0; j < Hh / 2; ++j) acc += hidden[(size_t)m * (Hh / 2) + j] * W2[j * NCc + cidx];
  scores[i] = acc;
}

// ---------------- kernel 7: top-k over time per (batch, class) -------------
__global__ void topk_kernel(const float* __restrict__ scores,
                            float* __restrict__ out_vals,
                            int* __restrict__ out_idx) {
  const int i = threadIdx.x;
  if (i >= Bsz * NCc) return;
  const int b = i / NCc, cidx = i % NCc;
  bool taken[Tt];
  for (int t = 0; t < Tt; ++t) taken[t] = false;
  float sum = 0.f;
  for (int k = 0; k < Kk; ++k) {
    float best = -3.4e38f; int bi = 0;
    for (int t = 0; t < Tt; ++t) {
      if (taken[t]) continue;
      const float v = scores[(size_t)(b * Tt + t) * NCc + cidx];
      if (v > best) { best = v; bi = t; }
    }
    taken[bi] = true;
    sum += best;
    out_idx[(b * Kk + k) * NCc + cidx] = bi;
  }
  out_vals[b * NCc + cidx] = sum * (1.0f / Kk);
}

// ---------------- host-side launch -----------------------------------------
extern "C" void kernel_launch(void* const* d_in, const int* in_sizes, int n_in,
                              void* d_out, int out_size, void* d_ws, size_t ws_size,
                              hipStream_t stream) {
  const float* x        = (const float*)d_in[0];
  const int*   ei       = (const int*)  d_in[1];
  const float* eattr    = (const float*)d_in[2];
  const float* W_l      = (const float*)d_in[3];
  const float* b_l      = (const float*)d_in[4];
  const float* W_r      = (const float*)d_in[5];
  const float* b_r      = (const float*)d_in[6];
  const float* W_e      = (const float*)d_in[7];
  const float* att      = (const float*)d_in[8];
  const float* bias_out = (const float*)d_in[9];
  const float* W_ih     = (const float*)d_in[10];
  const float* W_hh     = (const float*)d_in[11];
  const float* b_ih     = (const float*)d_in[12];
  const float* b_hh     = (const float*)d_in[13];
  const float* W1       = (const float*)d_in[14];
  const float* b1       = (const float*)d_in[15];
  const float* W2       = (const float*)d_in[16];
  const float* b2       = (const float*)d_in[17];

  const int* e_src = ei;
  const int* e_dst = ei + Ee;

  // ---------------- workspace layout ----------------
  // f32 region first, then 32B-aligned f16 region (all counts multiples of 16)
  float* ws      = (float*)d_ws;
  float* xl      = ws;                                    // 49600*128 f32
  float* xr      = xl + (size_t)Nn * Cc;                  // 49600*128 f32
  float* gx_all  = xr + (size_t)Nn * Cc;                  // 800*768  f32
  float* hiddenb = gx_all + (size_t)Gg * 3 * Hh;          // 800*128  f32
  float* scoresb = hiddenb + (size_t)Gg * (Hh / 2);       // 2400     f32
  _Float16* h16  = (_Float16*)(scoresb + 2400);
  _Float16* xh        = h16;                              // 49600*64
  _Float16* pooled16  = xh + (size_t)Nn * INF_;           // 800*128
  _Float16* gruout16  = pooled16 + (size_t)Gg * Cc;       // 800*256
  _Float16* pW_l      = gruout16 + (size_t)Gg * Hh;       // 16 frags  * 512
  _Float16* pW_r      = pW_l + 16 * 512;                  // 16 frags  * 512
  _Float16* pW_ih     = pW_r + 16 * 512;                  // 192 frags * 512
  _Float16* pW_hh     = pW_ih + 192 * 512;                // 384 frags * 512
  _Float16* pW1       = pW_hh + 384 * 512;                // 64 frags  * 512

  float* out_vals = (float*)d_out;            // 48 floats [B,NC]
  int*   out_idx  = (int*)d_out + Bsz * NCc;  // 240 ints  [B,K,NC]

  // 0) one-time (per launch) f16 staging + WMMA-fragment packing
  cvt_f16_kernel<<<(Nn * INF_ / 4 + 255) / 256, 256, 0, stream>>>(x, xh, Nn * INF_ / 4);
  pack_w_kernel<<<2, 256, 0, stream>>>(W_l, pW_l, 8, 2, Cc, 0);        // [64,128] K-major
  pack_w_kernel<<<2, 256, 0, stream>>>(W_r, pW_r, 8, 2, Cc, 0);        // [64,128] K-major
  pack_w_kernel<<<24, 256, 0, stream>>>(W_ih, pW_ih, 48, 4, Cc, 1);    // [768,128] N-major
  pack_w_kernel<<<48, 256, 0, stream>>>(W_hh, pW_hh, 48, 8, Hh, 1);    // [768,256] N-major
  pack_w_kernel<<<8, 256, 0, stream>>>(W1, pW1, 8, 8, Hh / 2, 0);      // [256,128] K-major

  // 1) node transforms (WMMA): 49600 tiles (3100 m x 8 n x 2 mats) / 8 waves
  xform_kernel<<<6200, 256, 0, stream>>>(xh, pW_l, b_l, pW_r, b_r, xl, xr);

  // 2) per-graph GATv2 + mean pool, all in LDS (~70 KB/WG of 320 KB)
  const size_t gat_lds = (size_t)(2 * NODES * LDPAD + 560 + 2 * Cc + 4 * NODES) * sizeof(float);
  gat_kernel<<<Gg, 256, gat_lds, stream>>>(xl, xr, e_src, e_dst, eattr,
                                           W_e, att, bias_out, pooled16);

  // 3) batched input-side GRU gates (WMMA): 2400 tiles / 8 waves
  gx_kernel<<<300, 256, 0, stream>>>(pooled16, pW_ih, gx_all);

  // 4) GRU recurrence: single persistent workgroup, WMMA per step (~98 KB LDS)
  const size_t gru_lds = (size_t)(Bsz * 3 * Hh + 2 * Bsz * Hh) * sizeof(float)
                       + (size_t)(2 * Bsz * Hh) * sizeof(_Float16);
  gru_kernel<<<1, 256, gru_lds, stream>>>(gx_all, pW_hh, b_ih, b_hh, gruout16);

  // 5) classifier layer 1 (WMMA + fused bias/ReLU): 400 tiles / 8 waves
  hidden_kernel<<<50, 256, 0, stream>>>(gruout16, pW1, b1, hiddenb);

  // 6) classifier layer 2 (N=3, scalar)
  scores_kernel<<<(Bsz * Tt * NCc + 255) / 256, 256, 0, stream>>>(hiddenb, W2, b2, scoresb);

  // 7) top-k over time + mean
  topk_kernel<<<1, 64, 0, stream>>>(scoresb, out_vals, out_idx);
}
